// EfficientAttention_90915867722244
// MI455X (gfx1250) — matrix operand order repaired and checked by reading, not compiled, not run on hardware
//
#include <hip/hip_runtime.h>

typedef __attribute__((ext_vector_type(16))) _Float16 v16h;
typedef __attribute__((ext_vector_type(8)))  _Float16 v8h;
typedef __attribute__((ext_vector_type(8)))  float    v8f;

constexpr int BATCH = 2;
constexpr int SEQ   = 2048;
constexpr int DIM   = 1024;
constexpr int NH    = 16;
constexpr int HD    = 64;
constexpr int MTOT  = BATCH * SEQ;   // 4096

// ---------------------------------------------------------------------------
// WMMA fragment helpers (wave32, 16x16x32 f16 -> f32)
// A 16x32 (MxK) f16: lane = row M (mod 16); lanes 0-15 hold K{0..7,16..23},
// lanes 16-31 hold K{8..15,24..31}; two contiguous 16B chunks per lane.
// B 32x16 (KxN) f16 loaded from a transposed (Bt[n][k]) LDS image:
// lane&15 = column N; lanes 0-15 hold K{0..7,16..23}, lanes 16-31 K{8..15,24..31}.
// C/D 16x16 f32: VGPR r -> row r + 8*(lane>>4), column lane&15.
// ---------------------------------------------------------------------------
__device__ inline v16h frag_a_lds(const _Float16* base, int stride_h) {
  const int lane = threadIdx.x & 31;
  const int hg = lane >> 4;
  const int m  = lane & 15;
  const _Float16* p = base + m * stride_h + hg * 8;
  v8h lo = *(const v8h*)(p);
  v8h hi = *(const v8h*)(p + 16);
  v16h r;
#pragma unroll
  for (int i = 0; i < 8; ++i) { r[i] = lo[i]; r[i + 8] = hi[i]; }
  return r;
}

__device__ inline v16h frag_bT_lds(const _Float16* base, int stride_h) {
  const int lane = threadIdx.x & 31;
  const int kh = lane >> 4;
  const int n  = lane & 15;
  const _Float16* p = base + n * stride_h + kh * 8;
  v8h lo = *(const v8h*)(p);
  v8h hi = *(const v8h*)(p + 16);
  v16h r;
#pragma unroll
  for (int i = 0; i < 8; ++i) { r[i] = lo[i]; r[i + 8] = hi[i]; }
  return r;
}

__device__ inline v8f wmma_f16(v16h a, v16h b, v8f c) {
  return __builtin_amdgcn_wmma_f32_16x16x32_f16(false, a, false, b,
                                                (short)0, c, false, false);
}

// ---------------------------------------------------------------------------
// fp32 -> fp16 convert
// ---------------------------------------------------------------------------
__global__ void cvt_f32_to_f16_kernel(const float* __restrict__ in,
                                      _Float16* __restrict__ out, int n) {
  int i = blockIdx.x * blockDim.x + threadIdx.x;
  const int stride = gridDim.x * blockDim.x;
  for (; i < n; i += stride) out[i] = (_Float16)in[i];
}

// ---------------------------------------------------------------------------
// fp32 [k][n] -> fp16 transposed [n][k] (for WMMA B-fragment loading)
// ---------------------------------------------------------------------------
__global__ void transpose_f32_to_f16_kernel(const float* __restrict__ w,
                                            _Float16* __restrict__ wt) {
  __shared__ float tile[32][33];
  const int bx = blockIdx.x * 32;
  const int by = blockIdx.y * 32;
  const int tx = threadIdx.x;
  for (int i = threadIdx.y; i < 32; i += 8)
    tile[i][tx] = w[(size_t)(by + i) * DIM + bx + tx];
  __syncthreads();
  for (int i = threadIdx.y; i < 32; i += 8)
    wt[(size_t)(bx + i) * DIM + by + tx] = (_Float16)tile[tx][i];
}

// ---------------------------------------------------------------------------
// WMMA GEMM: C[M][N] = X[M][K] * W[K][N] + bias   (K = N = DIM, M = MTOT)
// block tile 128x128, 8 waves, wave tile 32x64, K-step 32, software-pipelined
// global->LDS staging (next tile's loads overlap current tile's WMMAs).
// EPI 0: f16 out, [B,H,S,HD]      (Q, K)
// EPI 1: f32 out, row-major       (final projection -> d_out)
// EPI 2: f16 out, [B,H,HD,S]      (V pre-transposed; packed b128 stores)
// ---------------------------------------------------------------------------
template <int EPI>
__global__ __launch_bounds__(256) void gemm_kernel(
    const _Float16* __restrict__ X,   // [MTOT][DIM]
    const _Float16* __restrict__ WT,  // [DIM(n)][DIM(k)] (transposed weights)
    const float* __restrict__ bias,   // [DIM]
    _Float16* __restrict__ out_h, float* __restrict__ out_f) {
  constexpr int LSTR = 40;  // halves per LDS row (32 data + 8 pad) = 80B
  __shared__ _Float16 xs[128 * LSTR];
  __shared__ _Float16 wsm[128 * LSTR];

  const int n0 = blockIdx.x * 128;
  const int m0 = blockIdx.y * 128;
  const int tid = threadIdx.x;
  const int wave = tid >> 5;
  const int wm = (wave & 3) * 32;
  const int wn = (wave >> 2) * 64;

  v8f acc[2][4];
#pragma unroll
  for (int i = 0; i < 2; ++i)
#pragma unroll
    for (int j = 0; j < 4; ++j)
#pragma unroll
      for (int e = 0; e < 8; ++e) acc[i][j][e] = 0.0f;

  const int row = tid >> 1;
  const int seg = (tid & 1) * 16;
  const _Float16* xrow = X + (size_t)(m0 + row) * DIM + seg;
  const _Float16* wrow = WT + (size_t)(n0 + row) * DIM + seg;

  // prologue: fetch first k-tile into registers
  v8h x0 = *(const v8h*)(xrow);
  v8h x1 = *(const v8h*)(xrow + 8);
  v8h w0 = *(const v8h*)(wrow);
  v8h w1 = *(const v8h*)(wrow + 8);

  for (int k0 = 0; k0 < DIM; k0 += 32) {
    {
      v8h* dx = (v8h*)(xs + row * LSTR + seg);
      dx[0] = x0; dx[1] = x1;
      v8h* dw = (v8h*)(wsm + row * LSTR + seg);
      dw[0] = w0; dw[1] = w1;
    }
    __syncthreads();
    if (k0 + 32 < DIM) {  // issue next tile's global loads before computing
      x0 = *(const v8h*)(xrow + k0 + 32);
      x1 = *(const v8h*)(xrow + k0 + 40);
      w0 = *(const v8h*)(wrow + k0 + 32);
      w1 = *(const v8h*)(wrow + k0 + 40);
    }
    v16h af[2], bf[4];
#pragma unroll
    for (int i = 0; i < 2; ++i) af[i] = frag_a_lds(xs + (wm + i * 16) * LSTR, LSTR);
#pragma unroll
    for (int j = 0; j < 4; ++j) bf[j] = frag_bT_lds(wsm + (wn + j * 16) * LSTR, LSTR);
#pragma unroll
    for (int i = 0; i < 2; ++i)
#pragma unroll
      for (int j = 0; j < 4; ++j) acc[i][j] = wmma_f16(af[i], bf[j], acc[i][j]);
    __syncthreads();
  }

  const int lane = tid & 31;
  const int cn = lane & 15;
  const int rb = (lane >> 4) * 8;
#pragma unroll
  for (int i = 0; i < 2; ++i) {
#pragma unroll
    for (int j = 0; j < 4; ++j) {
      const int n = n0 + wn + j * 16 + cn;
      const float bval = bias[n];
      const int mb = m0 + wm + i * 16 + rb;     // first of 8 consecutive rows
      if constexpr (EPI == 2) {
        // V: write [B,H,HD,S]; 8 consecutive rows = 8 consecutive s -> b128
        const int b = mb >> 11;
        const int s0 = mb & (SEQ - 1);
        const int h = n >> 6;
        const int d = n & (HD - 1);
        v8h pk;
#pragma unroll
        for (int r = 0; r < 8; ++r) pk[r] = (_Float16)(acc[i][j][r] + bval);
        *(v8h*)(out_h + (((size_t)b * NH + h) * HD + d) * SEQ + s0) = pk;
      } else {
#pragma unroll
        for (int r = 0; r < 8; ++r) {
          const int m = mb + r;
          const float v = acc[i][j][r] + bval;
          if constexpr (EPI == 0) {
            const int b = m >> 11;        // / SEQ
            const int s = m & (SEQ - 1);
            const int h = n >> 6;         // / HD
            const int d = n & (HD - 1);
            out_h[(((size_t)b * NH + h) * SEQ + s) * HD + d] = (_Float16)v;
          } else {
            out_f[(size_t)m * DIM + n] = v;
          }
        }
      }
    }
  }
}

// ---------------------------------------------------------------------------
// Flash attention, causal. grid = (B*H, SEQ/128). 8 waves, each wave owns
// 16 query rows. K staged row-major, V arrives pre-transposed [B,H,HD,S]
// (pure vector b128 staging), P re-shaped D-frag -> A-frag via per-wave LDS.
// ---------------------------------------------------------------------------
__global__ __launch_bounds__(256) void attn_kernel(
    const _Float16* __restrict__ Q, const _Float16* __restrict__ Km,
    const _Float16* __restrict__ Vt, _Float16* __restrict__ O) {
  constexpr int KSTR = 72;   // 64 + 8 pad halves (144B rows)
  constexpr int VSTR = 136;  // 128 + 8 pad halves (272B rows)
  constexpr int PSTR = 72;   // per-wave P chunk 16x64
  __shared__ _Float16 kbuf[128 * KSTR];
  __shared__ _Float16 vbuf[HD * VSTR];
  __shared__ _Float16 pbuf[8 * 16 * PSTR];

  const int bh = blockIdx.x;
  const int b = bh >> 4;       // / NH
  const int h = bh & 15;
  const int by = blockIdx.y;
  const int tid = threadIdx.x;
  const int wave = tid >> 5;
  const int lane = tid & 31;
  const int hg = lane >> 4;
  const int lm = lane & 15;
  const int q0 = by * 128 + wave * 16;

  const size_t hoff = ((size_t)b * NH + h) * SEQ * HD;  // == HD*SEQ per head
  const _Float16* Qh = Q + hoff;
  const _Float16* Kh = Km + hoff;
  const _Float16* Vh = Vt + hoff;   // [HD][SEQ] within head

  // Q A-fragments (16 rows x 64 hd) -> 2 frags, loaded directly from global
  v16h aq[2];
#pragma unroll
  for (int c = 0; c < 2; ++c) {
    const _Float16* p = Qh + (size_t)(q0 + lm) * HD + c * 32 + hg * 8;
    v8h lo = *(const v8h*)(p);
    v8h hi = *(const v8h*)(p + 16);
#pragma unroll
    for (int i = 0; i < 8; ++i) { aq[c][i] = lo[i]; aq[c][i + 8] = hi[i]; }
  }

  v8f oacc[4];
  float mrow[8], lrow[8];
#pragma unroll
  for (int t = 0; t < 4; ++t)
#pragma unroll
    for (int e = 0; e < 8; ++e) oacc[t][e] = 0.0f;
#pragma unroll
  for (int r = 0; r < 8; ++r) { mrow[r] = -1e30f; lrow[r] = 0.0f; }

  const float C2 = 0.125f * 1.44269504088896f;  // HD^-0.5 * log2(e)
  _Float16* pw = pbuf + wave * 16 * PSTR;

  const int srow = tid >> 1;        // K staging row 0..127
  const int sseg = (tid & 1) * 32;  // K half-row segment
  const int vd   = tid >> 2;        // V staging row (hd) 0..63
  const int vseg = (tid & 3) * 32;  // V quarter-row segment

  for (int kb = 0; kb <= by; ++kb) {
    // stage K block row-major; V block rows directly from pre-transposed Vt
    {
      const v8h* sk = (const v8h*)(Kh + (size_t)(kb * 128 + srow) * HD + sseg);
      v8h k0v = sk[0], k1v = sk[1], k2v = sk[2], k3v = sk[3];
      v8h* dk = (v8h*)(kbuf + srow * KSTR + sseg);
      dk[0] = k0v; dk[1] = k1v; dk[2] = k2v; dk[3] = k3v;
      const v8h* sv = (const v8h*)(Vh + (size_t)vd * SEQ + kb * 128 + vseg);
      v8h v0v = sv[0], v1v = sv[1], v2v = sv[2], v3v = sv[3];
      v8h* dv = (v8h*)(vbuf + vd * VSTR + vseg);
      dv[0] = v0v; dv[1] = v1v; dv[2] = v2v; dv[3] = v3v;
    }
    __syncthreads();

    // scores for 128 keys: 8 tiles of 16 columns, K-dim = HD in 2 chunks
    v8f sacc[8];
#pragma unroll
    for (int t = 0; t < 8; ++t) {
#pragma unroll
      for (int e = 0; e < 8; ++e) sacc[t][e] = 0.0f;
#pragma unroll
      for (int c = 0; c < 2; ++c) {
        v16h bfr = frag_bT_lds(kbuf + (t * 16) * KSTR + c * 32, KSTR);
        sacc[t] = wmma_f16(aq[c], bfr, sacc[t]);
      }
    }

    // scale + causal mask (diagonal block only)
    const bool diag = (kb == by);
    float pm[8];
#pragma unroll
    for (int r = 0; r < 8; ++r) pm[r] = -1e30f;
#pragma unroll
    for (int t = 0; t < 8; ++t) {
      const int col = kb * 128 + t * 16 + lm;
#pragma unroll
      for (int r = 0; r < 8; ++r) {
        float sc = sacc[t][r] * C2;
        const int qg = q0 + r + hg * 8;
        if (diag && col > qg) sc = -1e30f;
        sacc[t][r] = sc;
        pm[r] = fmaxf(pm[r], sc);
      }
    }
#pragma unroll
    for (int off = 1; off < 16; off <<= 1)
#pragma unroll
      for (int r = 0; r < 8; ++r) pm[r] = fmaxf(pm[r], __shfl_xor(pm[r], off, 32));

    float alpha[8];
#pragma unroll
    for (int r = 0; r < 8; ++r) {
      const float mn = fmaxf(mrow[r], pm[r]);
      alpha[r] = exp2f(mrow[r] - mn);
      mrow[r] = mn;
      lrow[r] *= alpha[r];
    }
#pragma unroll
    for (int t = 0; t < 4; ++t)
#pragma unroll
      for (int r = 0; r < 8; ++r) oacc[t][r] *= alpha[r];

    float psum[8];
#pragma unroll
    for (int r = 0; r < 8; ++r) psum[r] = 0.0f;

    // keys in two 64-wide chunks: materialize P in per-wave LDS, then PV
#pragma unroll
    for (int hc = 0; hc < 2; ++hc) {
#pragma unroll
      for (int t = 0; t < 4; ++t) {
        const int tt = hc * 4 + t;
#pragma unroll
        for (int r = 0; r < 8; ++r) {
          const float p = exp2f(sacc[tt][r] - mrow[r]);
          psum[r] += p;
          pw[(r + hg * 8) * PSTR + t * 16 + lm] = (_Float16)p;
        }
      }
#pragma unroll
      for (int kc = 0; kc < 2; ++kc) {
        v16h ap = frag_a_lds(pw + kc * 32, PSTR);
#pragma unroll
        for (int nt = 0; nt < 4; ++nt) {
          v16h bvf = frag_bT_lds(vbuf + (nt * 16) * VSTR + hc * 64 + kc * 32, VSTR);
          oacc[nt] = wmma_f16(ap, bvf, oacc[nt]);
        }
      }
    }
#pragma unroll
    for (int off = 1; off < 16; off <<= 1)
#pragma unroll
      for (int r = 0; r < 8; ++r) psum[r] += __shfl_xor(psum[r], off, 32);
#pragma unroll
    for (int r = 0; r < 8; ++r) lrow[r] += psum[r];

    __syncthreads();
  }

  // finalize: divide by row sums, write f16 [B,S,D]
#pragma unroll
  for (int r = 0; r < 8; ++r) {
    const float inv = 1.0f / lrow[r];
    const int qg = q0 + r + hg * 8;
    _Float16* dst = O + ((size_t)b * SEQ + qg) * DIM + h * HD;
#pragma unroll
    for (int nt = 0; nt < 4; ++nt)
      dst[nt * 16 + lm] = (_Float16)(oacc[nt][r] * inv);
  }
}

// ---------------------------------------------------------------------------
extern "C" void kernel_launch(void* const* d_in, const int* in_sizes, int n_in,
                              void* d_out, int out_size, void* d_ws, size_t ws_size,
                              hipStream_t stream) {
  (void)in_sizes; (void)n_in; (void)out_size; (void)ws_size;
  const float* x  = (const float*)d_in[0];
  const float* wq = (const float*)d_in[1];
  const float* bq = (const float*)d_in[2];
  const float* wk = (const float*)d_in[3];
  const float* bk = (const float*)d_in[4];
  const float* wv = (const float*)d_in[5];
  const float* bv = (const float*)d_in[6];
  const float* wo = (const float*)d_in[7];
  const float* bo = (const float*)d_in[8];
  float* out = (float*)d_out;

  // workspace layout (f16): ~50.3 MB total
  _Float16* p = (_Float16*)d_ws;
  _Float16* xh  = p; p += (size_t)MTOT * DIM;
  _Float16* wqT = p; p += (size_t)DIM * DIM;
  _Float16* wkT = p; p += (size_t)DIM * DIM;
  _Float16* wvT = p; p += (size_t)DIM * DIM;
  _Float16* woT = p; p += (size_t)DIM * DIM;
  _Float16* qh  = p; p += (size_t)MTOT * DIM;
  _Float16* kh  = p; p += (size_t)MTOT * DIM;
  _Float16* vh  = p; p += (size_t)MTOT * DIM;   // [B,H,HD,S] (pre-transposed)
  _Float16* ah  = p; p += (size_t)MTOT * DIM;

  cvt_f32_to_f16_kernel<<<1024, 256, 0, stream>>>(x, xh, MTOT * DIM);
  dim3 tb(32, 8), tg(DIM / 32, DIM / 32);
  transpose_f32_to_f16_kernel<<<tg, tb, 0, stream>>>(wq, wqT);
  transpose_f32_to_f16_kernel<<<tg, tb, 0, stream>>>(wk, wkT);
  transpose_f32_to_f16_kernel<<<tg, tb, 0, stream>>>(wv, wvT);
  transpose_f32_to_f16_kernel<<<tg, tb, 0, stream>>>(wo, woT);

  dim3 gg(DIM / 128, MTOT / 128);  // (8, 32)
  gemm_kernel<0><<<gg, 256, 0, stream>>>(xh, wqT, bq, qh, nullptr);
  gemm_kernel<0><<<gg, 256, 0, stream>>>(xh, wkT, bk, kh, nullptr);
  gemm_kernel<2><<<gg, 256, 0, stream>>>(xh, wvT, bv, vh, nullptr);

  dim3 ga(BATCH * NH, SEQ / 128);  // (32, 16)
  attn_kernel<<<ga, 256, 0, stream>>>(qh, kh, vh, ah);

  gemm_kernel<1><<<gg, 256, 0, stream>>>(ah, woT, bo, nullptr, out);
}
